// MixtureOfExpertsLayer_80736795231131
// MI455X (gfx1250) — matrix operand order repaired and checked by reading, not compiled, not run on hardware
//
#include <hip/hip_runtime.h>
#include <hip/hip_bf16.h>

#define DPAR 1024
#define HPAR 4096
#define EPAR 8
#define TM   32          // tokens per FFN tile
#define NCOL 256         // output columns per wave in phase 2

typedef __attribute__((ext_vector_type(16))) __bf16 v16bf;
typedef __attribute__((ext_vector_type(8)))  __bf16 v8bf;
typedef __attribute__((ext_vector_type(4)))  __bf16 v4bf;
typedef __attribute__((ext_vector_type(8)))  float  v8f;

__device__ __forceinline__ v16bf cat8(v8bf a, v8bf b) {
    return __builtin_shufflevector(a, b, 0,1,2,3,4,5,6,7,8,9,10,11,12,13,14,15);
}

__device__ __forceinline__ float gelu_exact(float v) {
    return 0.5f * v * (1.0f + erff(v * 0.70710678118654752440f));
}

// Load 16 contiguous K-elements of a B operand (row-major weights) as bf16.
template <typename WT>
__device__ __forceinline__ v16bf load_b16(const WT* p) {
    if constexpr (sizeof(WT) == 2) {
        const v8bf lo = *(const v8bf*)p;
        const v8bf hi = *(const v8bf*)(p + 8);
        return cat8(lo, hi);
    } else {
        v16bf b;
#pragma unroll
        for (int q = 0; q < 4; ++q) {
            const float4 f4 = *(const float4*)((const float*)p + 4 * q);
            b[4*q+0] = (__bf16)f4.x; b[4*q+1] = (__bf16)f4.y;
            b[4*q+2] = (__bf16)f4.z; b[4*q+3] = (__bf16)f4.w;
        }
        return b;
    }
}

// ---------------------------------------------------------------------------
// f32 -> bf16 bulk conversion (4 elements / thread)
// ---------------------------------------------------------------------------
__global__ __launch_bounds__(256, 1)
void cvt_f32_bf16(const float* __restrict__ src, __bf16* __restrict__ dst,
                  long long n4)
{
    const long long i = (long long)blockIdx.x * blockDim.x + threadIdx.x;
    if (i >= n4) return;
    const float4 f = ((const float4*)src)[i];
    v4bf o; o[0] = (__bf16)f.x; o[1] = (__bf16)f.y;
            o[2] = (__bf16)f.z; o[3] = (__bf16)f.w;
    *(v4bf*)(dst + 4 * i) = o;
}

// ---------------------------------------------------------------------------
// Kernel 1: gating. One wave32 per token.
// ---------------------------------------------------------------------------
__global__ __launch_bounds__(256, 1)
void moe_gate(const float* __restrict__ x, const float* __restrict__ gate_w,
              const float* __restrict__ b2, float* __restrict__ cw,
              int* __restrict__ counts, int* __restrict__ lists,
              float* __restrict__ out, int T)
{
    const int lane = threadIdx.x & 31;
    const int wv   = threadIdx.x >> 5;
    const int t    = blockIdx.x * 8 + wv;
    if (t >= T) return;

    float acc[EPAR];
#pragma unroll
    for (int e = 0; e < EPAR; ++e) acc[e] = 0.0f;

    const float* xr = x + (size_t)t * DPAR;
    for (int d = lane; d < DPAR; d += 32) {
        const float xv = xr[d];
#pragma unroll
        for (int e = 0; e < EPAR; ++e) acc[e] += xv * gate_w[e * DPAR + d];
    }
#pragma unroll
    for (int e = 0; e < EPAR; ++e) {
#pragma unroll
        for (int off = 16; off > 0; off >>= 1)
            acc[e] += __shfl_xor(acc[e], off, 32);
    }

    float mx = acc[0];
#pragma unroll
    for (int e = 1; e < EPAR; ++e) mx = fmaxf(mx, acc[e]);
    float p[EPAR]; float s = 0.0f;
#pragma unroll
    for (int e = 0; e < EPAR; ++e) { p[e] = __expf(acc[e] - mx); s += p[e]; }
    const float inv = 1.0f / s;
#pragma unroll
    for (int e = 0; e < EPAR; ++e) p[e] *= inv;

    int i1 = 0;
#pragma unroll
    for (int e = 1; e < EPAR; ++e) if (p[e] > p[i1]) i1 = e;
    int i2 = (i1 == 0) ? 1 : 0;
#pragma unroll
    for (int e = 0; e < EPAR; ++e) if (e != i1 && p[e] > p[i2]) i2 = e;
    const float norm = p[i1] + p[i2];
    const float w1v = p[i1] / norm, w2v = p[i2] / norm;

    float cwl[EPAR];
#pragma unroll
    for (int e = 0; e < EPAR; ++e)
        cwl[e] = (e == i1) ? w1v : ((e == i2) ? w2v : 0.0f);

    if (lane == 0) {
#pragma unroll
        for (int e = 0; e < EPAR; ++e) cw[(size_t)t * EPAR + e] = cwl[e];
        int pos1 = atomicAdd(counts + i1, 1); lists[(size_t)i1 * T + pos1] = t;
        int pos2 = atomicAdd(counts + i2, 1); lists[(size_t)i2 * T + pos2] = t;
    }

    for (int dd = lane; dd < DPAR; dd += 32) {
        float s2 = 0.0f;
#pragma unroll
        for (int e = 0; e < EPAR; ++e) s2 += cwl[e] * b2[e * DPAR + dd];
        out[(size_t)t * DPAR + dd] = s2;
    }
}

// ---------------------------------------------------------------------------
// Kernel 2: fused per-expert FFN, 32-token tile, 8 waves, dynamic LDS.
//   per 32-hidden chunk:
//     phase1: h(32x32) = gelu(X(32x1024) @ w1_chunk^T + b1)
//             (4 subtiles, K split over 2 waves each, LDS reduce)
//     phase2: acc(16x256 per wave) += h @ w2_chunk^T
//   Operands are software-pipelined (next-iteration A/B loads issued before
//   the current WMMA) so waits relax from full drains to loadcnt/dscnt <= 2.
//   epilogue: out[token][n] += cw * acc  (global_atomic_add_f32)
// LDS: X 64KB bf16 | pbuf 8KB | h 2KB | idx/cw
// ---------------------------------------------------------------------------
template <typename WT>
__global__ __launch_bounds__(256, 1)
void moe_ffn(const WT* __restrict__ x, const WT* __restrict__ w1,
             const float* __restrict__ b1, const WT* __restrict__ w2,
             const float* __restrict__ cw, const int* __restrict__ counts,
             const int* __restrict__ lists, float* __restrict__ out,
             int T, int tilesPerE)
{
    extern __shared__ char smem[];
    __bf16* lds_x = (__bf16*)smem;                              // 65536 B
    float*  pbuf  = (float*)(smem + TM * DPAR * 2);             //  8192 B
    __bf16* lds_h = (__bf16*)(smem + TM * DPAR * 2 + 8192);     //  2048 B
    int*    s_idx = (int*)  (smem + TM * DPAR * 2 + 8192 + 2048);
    float*  s_cw  = (float*)(smem + TM * DPAR * 2 + 8192 + 2048 + TM * 4);

    const int e    = blockIdx.x / tilesPerE;
    const int tile = blockIdx.x % tilesPerE;
    const int cnt  = counts[e];
    if (tile * TM >= cnt) return;
    const int nvalid = min(TM, cnt - tile * TM);

    const int tid    = threadIdx.x;
    const int lane   = tid & 31;
    const int wv     = tid >> 5;
    const int m_lane = lane & 15;
    const int half   = lane >> 4;

    if (tid < TM) {
        const int base = e * T + tile * TM;
        const int t = (tid < nvalid) ? lists[base + tid] : lists[base];
        s_idx[tid] = t;
        s_cw[tid]  = (tid < nvalid) ? cw[(size_t)t * EPAR + e] : 0.0f;
    }
    __syncthreads();

    // ---- gather X tile (TM x 1024) into LDS as bf16 ----
    if constexpr (sizeof(WT) == 2) {
#pragma unroll
        for (int it = 0; it < 16; ++it) {
            const int linear = tid + it * 256;
            const int row = linear >> 7;              // 128 uint4 per row
            const int c8  = (linear & 127) << 3;
            const uint4 u = *(const uint4*)((const __bf16*)x
                              + (size_t)s_idx[row] * DPAR + c8);
            *(uint4*)(lds_x + row * DPAR + c8) = u;
        }
    } else {
#pragma unroll
        for (int it = 0; it < 32; ++it) {
            const int linear = tid + it * 256;
            const int row = linear >> 8;
            const int c4  = (linear & 255) << 2;
            const float4 f4 = *(const float4*)((const float*)x
                                + (size_t)s_idx[row] * DPAR + c4);
            v4bf o; o[0] = (__bf16)f4.x; o[1] = (__bf16)f4.y;
                    o[2] = (__bf16)f4.z; o[3] = (__bf16)f4.w;
            *(v4bf*)(lds_x + row * DPAR + c4) = o;
        }
    }
    __syncthreads();

    v8f acc[16];
#pragma unroll
    for (int j = 0; j < 16; ++j) acc[j] = (v8f){};

    const WT* w1e = w1 + (size_t)e * HPAR * DPAR;
    const WT* w2e = w2 + (size_t)e * DPAR * HPAR;

    // phase-1 subtile assignment: wave -> (mi, ni, k-group)
    const int st = wv & 3;
    const int mi = st & 1;            // token half (A rows)
    const int ni = st >> 1;           // hidden half (B cols)
    const int kg = wv >> 2;           // K group: k-steps kg, kg+2, ... (16 total)
    // phase-2 assignment: wave -> (token half, 256-column slice)
    const int mh = wv & 1;
    const int n0 = (wv >> 1) * NCOL;

    const __bf16* arow = lds_x + (16 * mi + m_lane) * DPAR + 16 * half;

    for (int hc = 0; hc < HPAR; hc += 32) {
        // ---------- phase 1 (A/B double-buffered) ----------
        const WT* brow = w1e + (size_t)(hc + 16 * ni + m_lane) * DPAR + 16 * half;
        v8f part = (v8f){};
        v8bf a_lo = *(const v8bf*)(arow + 32 * kg);
        v8bf a_hi = *(const v8bf*)(arow + 32 * kg + 8);
        v16bf bcur = load_b16(brow + 32 * kg);
#pragma unroll
        for (int jj = 0; jj < 16; ++jj) {
            const v16bf a = cat8(a_lo, a_hi);
            v8bf  al_n, ah_n;
            v16bf b_n;
            if (jj < 15) {                      // issue next-iter loads first
                const int sn = kg + (jj + 1) * 2;
                al_n = *(const v8bf*)(arow + 32 * sn);
                ah_n = *(const v8bf*)(arow + 32 * sn + 8);
                b_n  = load_b16(brow + 32 * sn);
            }
            part = __builtin_amdgcn_wmma_f32_16x16x32_bf16(
                false, a, false, bcur, (short)0, part, false, false);
            if (jj < 15) { a_lo = al_n; a_hi = ah_n; bcur = b_n; }
        }
#pragma unroll
        for (int v = 0; v < 8; ++v) pbuf[(wv * 32 + lane) * 8 + v] = part[v];
        __syncthreads();

        // ---------- reduce 2 partials + bias + gelu -> lds_h ----------
        if (wv < 4) {
            const int rmi = wv & 1, rni = wv >> 1;
            const float b1v = b1[e * HPAR + hc + 16 * rni + m_lane];
#pragma unroll
            for (int v = 0; v < 8; ++v) {
                float sv = pbuf[(wv * 32 + lane) * 8 + v]
                         + pbuf[((wv + 4) * 32 + lane) * 8 + v];
                sv = gelu_exact(sv + b1v);
                const int m = 16 * rmi + v + 8 * half;   // token row
                const int n = m_lane + 16 * rni;         // hidden col
                lds_h[m * 32 + n] = (__bf16)sv;
            }
        }
        __syncthreads();

        // ---------- phase 2 (B double-buffered; A loop-invariant) ----------
        const v8bf h_lo = *(const v8bf*)(lds_h + (16 * mh + m_lane) * 32 + 16 * half);
        const v8bf h_hi = *(const v8bf*)(lds_h + (16 * mh + m_lane) * 32 + 16 * half + 8);
        const v16bf ah = cat8(h_lo, h_hi);
        const WT* b2base = w2e + (size_t)(n0 + m_lane) * HPAR + hc + 16 * half;
        v16bf b2cur = load_b16(b2base);
#pragma unroll
        for (int j = 0; j < 16; ++j) {
            v16bf b2n;
            if (j < 15)
                b2n = load_b16(b2base + (size_t)(j + 1) * 16 * HPAR);
            acc[j] = __builtin_amdgcn_wmma_f32_16x16x32_bf16(
                false, ah, false, b2cur, (short)0, acc[j], false, false);
            if (j < 15) b2cur = b2n;
        }

        // prefetch next weight chunk (global_prefetch_b8)
        if (hc + 32 < HPAR) {
            __builtin_prefetch(w1e + (size_t)(hc + 32 + 16 * ni + m_lane) * DPAR
                                   + 16 * half, 0, 1);
            __builtin_prefetch(w2e + (size_t)(n0 + m_lane) * HPAR
                                   + hc + 32 + 16 * half, 0, 1);
        }
        __syncthreads();
    }

    // ---------- epilogue ----------
#pragma unroll
    for (int j = 0; j < 16; ++j) {
#pragma unroll
        for (int v = 0; v < 8; ++v) {
            const int m = 16 * mh + v + 8 * half;
            if (m < nvalid) {
                const int n = n0 + 16 * j + m_lane;
                atomicAdd(out + (size_t)s_idx[m] * DPAR + n, acc[j][v] * s_cw[m]);
            }
        }
    }
}

// ---------------------------------------------------------------------------
extern "C" void kernel_launch(void* const* d_in, const int* in_sizes, int n_in,
                              void* d_out, int out_size, void* d_ws, size_t ws_size,
                              hipStream_t stream)
{
    const float* x      = (const float*)d_in[0];
    const float* gate_w = (const float*)d_in[1];
    const float* w1     = (const float*)d_in[2];
    const float* b1     = (const float*)d_in[3];
    const float* w2     = (const float*)d_in[4];
    const float* b2     = (const float*)d_in[5];
    float* out = (float*)d_out;

    const int T = in_sizes[0] / DPAR;                 // 8192 tokens
    const size_t nW = (size_t)EPAR * HPAR * DPAR;     // w1 (== w2) elements
    const size_t nX = (size_t)T * DPAR;

    // workspace layout
    char* ws = (char*)d_ws;
    size_t off = 0;
    float* cw   = (float*)(ws + off); off += (size_t)T * EPAR * sizeof(float);
    int* counts = (int*)(ws + off);   off += 128;
    int* lists  = (int*)(ws + off);   off += (size_t)EPAR * T * sizeof(int);
    off = (off + 255) & ~(size_t)255;
    __bf16* xb  = (__bf16*)(ws + off); off += nX * 2;
    __bf16* w1b = (__bf16*)(ws + off); off += nW * 2;
    __bf16* w2b = (__bf16*)(ws + off); off += nW * 2;
    const bool fast = (ws_size >= off);

    hipMemsetAsync(counts, 0, 128, stream);

    const int smemBytes = TM * DPAR * 2 + 8192 + 2048 + TM * 4 + TM * 4;
    const int tilesPerE = (T + TM - 1) / TM;          // 256

    moe_gate<<<(T + 7) / 8, 256, 0, stream>>>(x, gate_w, b2, cw, counts,
                                              lists, out, T);

    if (fast) {
        const long long n4w = (long long)(nW / 4);
        const long long n4x = (long long)(nX / 4);
        cvt_f32_bf16<<<(int)((n4w + 255) / 256), 256, 0, stream>>>(w1, w1b, n4w);
        cvt_f32_bf16<<<(int)((n4w + 255) / 256), 256, 0, stream>>>(w2, w2b, n4w);
        cvt_f32_bf16<<<(int)((n4x + 255) / 256), 256, 0, stream>>>(x,  xb,  n4x);
        moe_ffn<__bf16><<<EPAR * tilesPerE, 256, smemBytes, stream>>>(
            xb, w1b, b1, w2b, cw, counts, lists, out, T, tilesPerE);
    } else {
        moe_ffn<float><<<EPAR * tilesPerE, 256, smemBytes, stream>>>(
            x, w1, b1, w2, cw, counts, lists, out, T, tilesPerE);
    }
}
